// QuantizedLinear_78958678770031
// MI455X (gfx1250) — compile-verified, matrix operand order checked by reading, not run on hardware
//
#include <hip/hip_runtime.h>

// ---------------------------------------------------------------------------
// QuantizedLinear: out = x @ (wq * scale)^T + bias
// GEMM M=8192 N=4096 K=4096.
// Fast path: pre-convert x (fp32->f16) and wq (int32->f16) into d_ws, then a
// TDM-staged, double-buffered f16 WMMA GEMM (128x256 block tile, 64x64 wave
// tile, BK=64). Fallback (ws too small): fused-conversion WMMA GEMM.
// ---------------------------------------------------------------------------

typedef __attribute__((ext_vector_type(16))) _Float16     v16h;
typedef __attribute__((ext_vector_type(8)))  _Float16     v8h;
typedef __attribute__((ext_vector_type(4)))  _Float16     v4h;
typedef __attribute__((ext_vector_type(8)))  float        v8f;
typedef __attribute__((ext_vector_type(4)))  unsigned int u32x4;
typedef __attribute__((ext_vector_type(8)))  int          i32x8;
typedef __attribute__((ext_vector_type(4)))  int          i32x4;

#define M_DIM 8192
#define N_DIM 4096
#define K_DIM 4096

// ---- fast-path tiling ----
#define BM 128
#define BN 256
#define BK 64
// TDM pads 4 dwords (16B) after every 32 dwords (128B = one 64-half row):
// LDS row stride = 144 B = 72 halfs -> 16-row fragment reads are conflict-free
#define LDS_ROW_HALFS 72

// ---------------------------------------------------------------------------
// TDM descriptor build + issue: load a (tile_rows x BK) f16 tile from a
// row-major [tensor_rows x K_DIM] f16 matrix into LDS (padded rows).
// D# encoding per CDNA5 ISA 8.3/8.4. 2D tensor: groups 2/3 zero.
// amdgpu-toolchain (clang-23) builtin: 6 args
//   (u32x4 g0, i32x8 g1, i32x4, i32x4, i32x8, i32 cpol)
// ---------------------------------------------------------------------------
__device__ __forceinline__ void tdm_load_tile_f16(unsigned lds_off,
                                                  const _Float16* gbase,
                                                  int row0, int k0,
                                                  int tensor_rows, int tile_rows)
{
    unsigned long long ga = (unsigned long long)(size_t)gbase
                          + ((unsigned long long)(unsigned)row0 * K_DIM + (unsigned)k0) * 2ull;
    u32x4 g0;
    g0.x = 1u;                                                   // count=1, is_restore=0
    g0.y = lds_off;                                              // lds_addr (bytes)
    g0.z = (unsigned)ga;                                         // global_addr[31:0]
    g0.w = ((unsigned)(ga >> 32) & 0x01FFFFFFu) | 0x80000000u;   // global_addr[56:32] | type=2

    const unsigned td0 = K_DIM;                  // tensor_dim0 (elements along K)
    const unsigned td1 = (unsigned)tensor_rows;  // tensor_dim1 (rows)
    const unsigned w0  = (1u << 16)              // data_size = 2 bytes
                       | (1u << 20)              // pad_enable
                       | (4u << 22)              // pad_interval: 32 dwords (128B row)
                       | (3u << 25);             // pad_amount: 4 dwords (16B)
    const unsigned w1  = (td0 & 0xFFFFu) << 16;                          // dim0[15:0]  @63:48
    const unsigned w2  = ((td0 >> 16) & 0xFFFFu) | ((td1 & 0xFFFFu) << 16); // dim0[31:16], dim1[15:0]
    const unsigned w3  = ((td1 >> 16) & 0xFFFFu) | ((unsigned)BK << 16);    // dim1[31:16], tile_dim0
    const unsigned w4  = (unsigned)tile_rows & 0xFFFFu;                  // tile_dim1 (tile_dim2=0)
    const unsigned w5  = K_DIM;                                          // dim0_stride[31:0]
    i32x8 g1 = { (int)w0, (int)w1, (int)w2, (int)w3, (int)w4, (int)w5, 0, 0 };
    i32x4 z4 = { 0, 0, 0, 0 };
    i32x8 z8 = { 0, 0, 0, 0, 0, 0, 0, 0 };
    __builtin_amdgcn_tensor_load_to_lds(g0, g1, z4, z4, z8, 0);
}

// ---------------------------------------------------------------------------
// Fast-path GEMM: f16 inputs, TDM staging, double-buffered LDS, WMMA f32 acc.
// 256 threads = 8 waves (2 M-slabs x 4 N-slabs), each wave owns 64x64 of C.
// ---------------------------------------------------------------------------
__global__ __launch_bounds__(256)
void qlinear_wmma_tdm(const _Float16* __restrict__ Xh,
                      const _Float16* __restrict__ Wh,
                      const float* __restrict__ scale_p,
                      const float* __restrict__ bias,
                      float* __restrict__ Out)
{
    __shared__ __align__(16) _Float16 shA[2][BM * LDS_ROW_HALFS];
    __shared__ __align__(16) _Float16 shB[2][BN * LDS_ROW_HALFS];

    const int tid  = threadIdx.x;
    const int lane = tid & 31;
    const int wave = tid >> 5;
    const int wm   = wave >> 2;        // 0..1  -> 64-row slab
    const int wn   = wave & 3;         // 0..3  -> 64-col slab

    const int block_m = blockIdx.y * BM;
    const int block_n = blockIdx.x * BN;

    const float scale = scale_p[0];

    v8f acc[4][4] = {};

    // ISA 16-bit A/B fragment layout inside a 32-wide K step:
    // lanes 0-15 hold K {0..7,16..23}, lanes 16-31 hold K {8..15,24..31}
    const int kofs = (lane & 16) ? 8 : 0;
    const int frow = lane & 15;        // fragment row (M for A, N for B)
    const int crow = (lane >> 4) * 8;  // C/D layout: lanes 16-31 are rows +8

    const unsigned ldsA0 = (unsigned)(size_t)&shA[0][0];
    const unsigned ldsA1 = (unsigned)(size_t)&shA[1][0];
    const unsigned ldsB0 = (unsigned)(size_t)&shB[0][0];
    const unsigned ldsB1 = (unsigned)(size_t)&shB[1][0];

    const int  nk     = K_DIM / BK;    // 64
    const bool issuer = (wave == 0);   // one wave drives the TDM (EXEC ignored)

    if (issuer) {
        tdm_load_tile_f16(ldsA0, Xh, block_m, 0, M_DIM, BM);
        tdm_load_tile_f16(ldsB0, Wh, block_n, 0, N_DIM, BN);
    }

    for (int t = 0; t < nk; ++t) {
        const int buf = t & 1;
        if (issuer) {
            if (t + 1 < nk) {
                // stage next K-tile into the spare buffer (freed by the
                // trailing barrier of iteration t-1), then wait for the
                // *current* pair (leave only the 2 just-issued outstanding)
                tdm_load_tile_f16(buf ? ldsA0 : ldsA1, Xh, block_m, (t + 1) * BK, M_DIM, BM);
                tdm_load_tile_f16(buf ? ldsB0 : ldsB1, Wh, block_n, (t + 1) * BK, N_DIM, BN);
                __builtin_amdgcn_s_wait_tensorcnt(2);
            } else {
                __builtin_amdgcn_s_wait_tensorcnt(0);
            }
        }
        __syncthreads();               // current buffers visible to all waves

        const _Float16* As = shA[buf];
        const _Float16* Bs = shB[buf];

#pragma unroll
        for (int s = 0; s < 2; ++s) {  // two 32-wide K steps inside BK=64
            v16h bfrag[4];
#pragma unroll
            for (int j = 0; j < 4; ++j) {
                const _Float16* p = Bs + (wn * 64 + j * 16 + frow) * LDS_ROW_HALFS + s * 32 + kofs;
                v8h lo = *(const v8h*)(p);
                v8h hi = *(const v8h*)(p + 16);
                bfrag[j] = __builtin_shufflevector(lo, hi,
                    0, 1, 2, 3, 4, 5, 6, 7, 8, 9, 10, 11, 12, 13, 14, 15);
            }
#pragma unroll
            for (int i = 0; i < 4; ++i) {
                const _Float16* p = As + (wm * 64 + i * 16 + frow) * LDS_ROW_HALFS + s * 32 + kofs;
                v8h lo = *(const v8h*)(p);
                v8h hi = *(const v8h*)(p + 16);
                v16h afrag = __builtin_shufflevector(lo, hi,
                    0, 1, 2, 3, 4, 5, 6, 7, 8, 9, 10, 11, 12, 13, 14, 15);
#pragma unroll
                for (int j = 0; j < 4; ++j) {
                    acc[i][j] = __builtin_amdgcn_wmma_f32_16x16x32_f16(
                        false, afrag, false, bfrag[j],
                        (short)0, acc[i][j], false, false);
                }
            }
        }
        __syncthreads();               // all waves done reading buf
    }

    // ---- epilogue: out = acc * scale + bias ----
#pragma unroll
    for (int i = 0; i < 4; ++i) {
        const int m0 = block_m + wm * 64 + i * 16 + crow;
#pragma unroll
        for (int j = 0; j < 4; ++j) {
            const int n  = block_n + wn * 64 + j * 16 + frow;
            const float bi = bias[n];
#pragma unroll
            for (int v = 0; v < 8; ++v) {
                Out[(size_t)(m0 + v) * N_DIM + n] = acc[i][j][v] * scale + bi;
            }
        }
    }
}

// ---------------------------------------------------------------------------
// Pre-pass conversions into workspace
// ---------------------------------------------------------------------------
__global__ __launch_bounds__(256)
void convert_f32_to_f16(const float* __restrict__ in, _Float16* __restrict__ out, int n4)
{
    const int i = blockIdx.x * 256 + threadIdx.x;
    if (i < n4) {
        const float4 f = ((const float4*)in)[i];
        v4h h;
        h.x = (_Float16)f.x; h.y = (_Float16)f.y;
        h.z = (_Float16)f.z; h.w = (_Float16)f.w;
        ((v4h*)out)[i] = h;
    }
}

__global__ __launch_bounds__(256)
void convert_i32_to_f16(const int* __restrict__ in, _Float16* __restrict__ out, int n4)
{
    const int i = blockIdx.x * 256 + threadIdx.x;
    if (i < n4) {
        const int4 q = ((const int4*)in)[i];
        v4h h;
        h.x = (_Float16)(float)q.x; h.y = (_Float16)(float)q.y;
        h.z = (_Float16)(float)q.z; h.w = (_Float16)(float)q.w;
        ((v4h*)out)[i] = h;
    }
}

// ---------------------------------------------------------------------------
// Fallback (ws too small): fused-conversion WMMA GEMM (128x128 tile, BK=32)
// ---------------------------------------------------------------------------
__global__ __launch_bounds__(256)
void qlinear_wmma_fused(const float* __restrict__ X,
                        const int*   __restrict__ Wq,
                        const float* __restrict__ scale_p,
                        const float* __restrict__ bias,
                        float*       __restrict__ Out)
{
    __shared__ __align__(16) _Float16 shA[128 * 40];
    __shared__ __align__(16) _Float16 shB[128 * 40];

    const int tid  = threadIdx.x;
    const int lane = tid & 31;
    const int wave = tid >> 5;
    const int wm   = wave >> 2;
    const int wn   = wave & 3;

    const int block_m = blockIdx.y * 128;
    const int block_n = blockIdx.x * 128;
    const float scale = scale_p[0];

    v8f acc[4][2] = {};

    const int lrow = tid >> 3;
    const int lcol = (tid & 7) * 4;
    const int kofs = (lane & 16) ? 8 : 0;
    const int frow = lane & 15;
    const int crow = (lane >> 4) * 8;

    for (int k0 = 0; k0 < K_DIM; k0 += 32) {
#pragma unroll
        for (int r = 0; r < 4; ++r) {
            const int row = lrow + 32 * r;
            const float4 a4 = *(const float4*)(X + (size_t)(block_m + row) * K_DIM + k0 + lcol);
            v4h ah;
            ah.x = (_Float16)a4.x; ah.y = (_Float16)a4.y;
            ah.z = (_Float16)a4.z; ah.w = (_Float16)a4.w;
            *(v4h*)(shA + row * 40 + lcol) = ah;

            const int4 w4 = *(const int4*)(Wq + (size_t)(block_n + row) * K_DIM + k0 + lcol);
            v4h bh;
            bh.x = (_Float16)(float)w4.x; bh.y = (_Float16)(float)w4.y;
            bh.z = (_Float16)(float)w4.z; bh.w = (_Float16)(float)w4.w;
            *(v4h*)(shB + row * 40 + lcol) = bh;
        }
        __syncthreads();

        v16h bfrag[2];
#pragma unroll
        for (int j = 0; j < 2; ++j) {
            const _Float16* p = shB + (wn * 32 + j * 16 + frow) * 40 + kofs;
            v8h lo = *(const v8h*)(p);
            v8h hi = *(const v8h*)(p + 16);
            bfrag[j] = __builtin_shufflevector(lo, hi,
                0, 1, 2, 3, 4, 5, 6, 7, 8, 9, 10, 11, 12, 13, 14, 15);
        }
#pragma unroll
        for (int i = 0; i < 4; ++i) {
            const _Float16* p = shA + (wm * 64 + i * 16 + frow) * 40 + kofs;
            v8h lo = *(const v8h*)(p);
            v8h hi = *(const v8h*)(p + 16);
            v16h afrag = __builtin_shufflevector(lo, hi,
                0, 1, 2, 3, 4, 5, 6, 7, 8, 9, 10, 11, 12, 13, 14, 15);
#pragma unroll
            for (int j = 0; j < 2; ++j) {
                acc[i][j] = __builtin_amdgcn_wmma_f32_16x16x32_f16(
                    false, afrag, false, bfrag[j], (short)0, acc[i][j], false, false);
            }
        }
        __syncthreads();
    }

#pragma unroll
    for (int i = 0; i < 4; ++i) {
        const int m0 = block_m + wm * 64 + i * 16 + crow;
#pragma unroll
        for (int j = 0; j < 2; ++j) {
            const int n  = block_n + wn * 32 + j * 16 + frow;
            const float bi = bias[n];
#pragma unroll
            for (int v = 0; v < 8; ++v) {
                Out[(size_t)(m0 + v) * N_DIM + n] = acc[i][j][v] * scale + bi;
            }
        }
    }
}

// ---------------------------------------------------------------------------
extern "C" void kernel_launch(void* const* d_in, const int* in_sizes, int n_in,
                              void* d_out, int out_size, void* d_ws, size_t ws_size,
                              hipStream_t stream) {
    const float* x     = (const float*)d_in[0];
    const int*   wq    = (const int*)d_in[1];
    const float* scale = (const float*)d_in[2];
    const float* bias  = (const float*)d_in[3];
    float*       out   = (float*)d_out;

    const size_t xh_bytes = (size_t)M_DIM * K_DIM * sizeof(_Float16);   // 64 MiB
    const size_t wh_bytes = (size_t)N_DIM * K_DIM * sizeof(_Float16);   // 32 MiB

    if (ws_size >= xh_bytes + wh_bytes) {
        _Float16* Xh = (_Float16*)d_ws;
        _Float16* Wh = (_Float16*)((char*)d_ws + xh_bytes);

        const int n4x = M_DIM * K_DIM / 4;
        const int n4w = N_DIM * K_DIM / 4;
        convert_f32_to_f16<<<(n4x + 255) / 256, 256, 0, stream>>>(x, Xh, n4x);
        convert_i32_to_f16<<<(n4w + 255) / 256, 256, 0, stream>>>(wq, Wh, n4w);

        dim3 grid(N_DIM / BN, M_DIM / BM);   // 16 x 64
        qlinear_wmma_tdm<<<grid, 256, 0, stream>>>(Xh, Wh, scale, bias, out);
    } else {
        dim3 grid(N_DIM / 128, M_DIM / 128); // 32 x 64
        qlinear_wmma_fused<<<grid, 256, 0, stream>>>(x, wq, scale, bias, out);
    }
}
